// MHSA_75909251990222
// MI455X (gfx1250) — compile-verified, matrix operand order checked by reading, not run on hardware
//
#include <hip/hip_runtime.h>

// ---------- types ----------
typedef __bf16 bf16;
typedef __bf16 bf16x16 __attribute__((ext_vector_type(16)));
typedef __bf16 bf16x4  __attribute__((ext_vector_type(4)));
typedef float  f32x8   __attribute__((ext_vector_type(8)));
typedef unsigned int u32x4 __attribute__((ext_vector_type(4)));

union FragU { bf16x16 v; u32x4 q[2]; };

// D = A*B + C, 16x16x32 bf16 -> f32
__device__ inline f32x8 wmma_bf16(bf16x16 a, bf16x16 b, f32x8 c) {
    return __builtin_amdgcn_wmma_f32_16x16x32_bf16(
        /*neg_a=*/false, a, /*neg_b=*/false, b,
        /*c_mod=*/(short)0, c, /*reuse_a=*/false, /*reuse_b=*/false);
}

// Load a 16x32 bf16 operand fragment (A layout; B^T rows use the same layout).
// lane l (l<16): row = row0+l, holds K = k0 + {0..7, 16..23}
// lane l (l>=16): row = row0+l-16, holds K = k0 + {8..15, 24..31}
// => two contiguous 16-byte loads per lane.
__device__ inline bf16x16 load_frag(const bf16* base, int row0, int ld, int k0, int lane) {
    int r  = lane & 15;
    int hi = lane >> 4;
    const bf16* p = base + (size_t)(row0 + r) * (size_t)ld + (size_t)(k0 + hi * 8);
    FragU u;
    u.q[0] = *(const u32x4*)(p);
    u.q[1] = *(const u32x4*)(p + 16);
    return u.v;
}

// ---------- constants ----------
#define HEADS    16
#define HEAD_DIM 64
#define TOKDIM   1024
#define INNER    1024          // HEADS*HEAD_DIM
#define SEQ      2048
#define BATCH    2
#define NTOK     (BATCH*SEQ)   // 4096

// ---------- kernel 1: fp32 -> bf16 convert ----------
__global__ void cvt_f32_bf16(const float* __restrict__ src, bf16* __restrict__ dst, int n4) {
    int i = blockIdx.x * blockDim.x + threadIdx.x;
    if (i < n4) {
        const float4 v = ((const float4*)src)[i];
        bf16x4 o;
        o.x = (bf16)v.x; o.y = (bf16)v.y; o.z = (bf16)v.z; o.w = (bf16)v.w;
        *(bf16x4*)(dst + (size_t)i * 4) = o;   // single 8-byte store
    }
}

// ---------- kernel 2: QKV GEMM  qkv[m, j] = sum_k x[m,k] * w_qkv[j,k] ----------
// scatter: j<1024 -> Q[b,h,n,d]; j<2048 -> K[b,h,n,d]; else V^T[b,h,d,n]
__device__ inline void qkv_scatter(bf16* Q, bf16* K, bf16* Vt, int m, int j, float val) {
    int b = m >> 11, n = m & 2047;
    int which = j >> 10, jj = j & 1023;
    int h = jj >> 6, d = jj & 63;
    bf16 v = (bf16)val;
    size_t bh = (size_t)(b * HEADS + h);
    if (which == 0)      Q [(bh * SEQ + n) * HEAD_DIM + d] = v;
    else if (which == 1) K [(bh * SEQ + n) * HEAD_DIM + d] = v;
    else                 Vt[(bh * HEAD_DIM + d) * SEQ + n] = v;
}

__global__ __launch_bounds__(256) void qkv_gemm(const bf16* __restrict__ xb,
                                                const bf16* __restrict__ wb,
                                                bf16* __restrict__ Q,
                                                bf16* __restrict__ K,
                                                bf16* __restrict__ Vt) {
    int lane = threadIdx.x & 31, wid = threadIdx.x >> 5;
    int wm = wid & 3, wn = wid >> 2;
    int m0 = blockIdx.y * 128 + wm * 32;   // 4 waves along M
    int n0 = blockIdx.x * 128 + wn * 64;   // 2 waves along N

    f32x8 acc[2][4] = {};
    for (int k0 = 0; k0 < TOKDIM; k0 += 32) {
        if (k0 + 32 < TOKDIM) {
            // per-lane line-ahead prefetch, WGP scope (pulled into all levels)
            __builtin_prefetch(xb + (size_t)(m0 + lane) * TOKDIM + k0 + 32, 0, 3);
            __builtin_prefetch(wb + (size_t)(n0 + 2 * lane) * TOKDIM + k0 + 32, 0, 3);
        }
        bf16x16 a0 = load_frag(xb, m0,      TOKDIM, k0, lane);
        bf16x16 a1 = load_frag(xb, m0 + 16, TOKDIM, k0, lane);
        bf16x16 b0 = load_frag(wb, n0,      TOKDIM, k0, lane);
        bf16x16 b1 = load_frag(wb, n0 + 16, TOKDIM, k0, lane);
        bf16x16 b2 = load_frag(wb, n0 + 32, TOKDIM, k0, lane);
        bf16x16 b3 = load_frag(wb, n0 + 48, TOKDIM, k0, lane);
        acc[0][0] = wmma_bf16(a0, b0, acc[0][0]);
        acc[0][1] = wmma_bf16(a0, b1, acc[0][1]);
        acc[0][2] = wmma_bf16(a0, b2, acc[0][2]);
        acc[0][3] = wmma_bf16(a0, b3, acc[0][3]);
        acc[1][0] = wmma_bf16(a1, b0, acc[1][0]);
        acc[1][1] = wmma_bf16(a1, b1, acc[1][1]);
        acc[1][2] = wmma_bf16(a1, b2, acc[1][2]);
        acc[1][3] = wmma_bf16(a1, b3, acc[1][3]);
    }
    int hi = lane >> 4, col = lane & 15;
    #pragma unroll
    for (int i = 0; i < 2; i++)
        #pragma unroll
        for (int t = 0; t < 4; t++)
            #pragma unroll
            for (int r = 0; r < 8; r++)
                qkv_scatter(Q, K, Vt, m0 + i * 16 + r + 8 * hi, n0 + t * 16 + col, acc[i][t][r]);
}

// ---------- kernel 3: flash attention, one wave per 16-query tile ----------
__global__ __launch_bounds__(256) void attn_kernel(const bf16* __restrict__ Q,
                                                   const bf16* __restrict__ K,
                                                   const bf16* __restrict__ Vt,
                                                   bf16* __restrict__ O) {
    __shared__ __align__(16) bf16 plds[8][16 * 32];   // 1KB per wave
    int lane = threadIdx.x & 31, wid = threadIdx.x >> 5;
    int hi = lane >> 4, col = lane & 15;
    int bh = blockIdx.x;              // 0..31
    int q0 = blockIdx.y * 128 + wid * 16;

    const bf16* Qb = Q  + (size_t)bh * SEQ * HEAD_DIM;
    const bf16* Kb = K  + (size_t)bh * SEQ * HEAD_DIM;
    const bf16* Vb = Vt + (size_t)bh * HEAD_DIM * SEQ;

    bf16x16 qa0 = load_frag(Qb, q0, HEAD_DIM, 0,  lane);
    bf16x16 qa1 = load_frag(Qb, q0, HEAD_DIM, 32, lane);

    f32x8 o0 = {}, o1 = {}, o2 = {}, o3 = {};
    float mrow[8], lrow[8];
    #pragma unroll
    for (int r = 0; r < 8; r++) { mrow[r] = -1e30f; lrow[r] = 0.0f; }

    const float scale = 0.125f;   // 1/sqrt(64)

    for (int kb = 0; kb < SEQ; kb += 32) {
        if (kb + 32 < SEQ) {
            // warm WGP$ one key-chunk ahead (shared by all 8 waves of the block)
            __builtin_prefetch(Kb + (size_t)(kb + 32 + lane) * HEAD_DIM, 0, 3);
            __builtin_prefetch(Vb + (size_t)(2 * lane) * SEQ + kb + 32, 0, 3);
        }
        // scores: two 16x16 tiles over 32 keys, K-dim = head_dim = 64
        bf16x16 k00 = load_frag(Kb, kb,      HEAD_DIM, 0,  lane);
        bf16x16 k01 = load_frag(Kb, kb,      HEAD_DIM, 32, lane);
        bf16x16 k10 = load_frag(Kb, kb + 16, HEAD_DIM, 0,  lane);
        bf16x16 k11 = load_frag(Kb, kb + 16, HEAD_DIM, 32, lane);
        f32x8 s0 = {}, s1 = {};
        s0 = wmma_bf16(qa0, k00, s0); s0 = wmma_bf16(qa1, k01, s0);
        s1 = wmma_bf16(qa0, k10, s1); s1 = wmma_bf16(qa1, k11, s1);

        // online softmax per row (row r+8*hi lives in 16 lanes of this half)
        float p0[8], p1[8];
        #pragma unroll
        for (int r = 0; r < 8; r++) {
            float a = s0[r] * scale, b = s1[r] * scale;
            float v = fmaxf(a, b);
            v = fmaxf(v, __shfl_xor(v, 1, 32));
            v = fmaxf(v, __shfl_xor(v, 2, 32));
            v = fmaxf(v, __shfl_xor(v, 4, 32));
            v = fmaxf(v, __shfl_xor(v, 8, 32));
            float mnew = fmaxf(mrow[r], v);
            float fac  = __expf(mrow[r] - mnew);
            mrow[r] = mnew;
            p0[r] = __expf(a - mnew);
            p1[r] = __expf(b - mnew);
            float rs = p0[r] + p1[r];
            rs += __shfl_xor(rs, 1, 32);
            rs += __shfl_xor(rs, 2, 32);
            rs += __shfl_xor(rs, 4, 32);
            rs += __shfl_xor(rs, 8, 32);
            lrow[r] = lrow[r] * fac + rs;
            o0[r] *= fac; o1[r] *= fac; o2[r] *= fac; o3[r] *= fac;
        }

        // repack P (16 rows x 32 keys) from C layout into A layout via LDS
        #pragma unroll
        for (int r = 0; r < 8; r++) {
            int row = r + 8 * hi;
            plds[wid][row * 32 + col]      = (bf16)p0[r];
            plds[wid][row * 32 + col + 16] = (bf16)p1[r];
        }
        __syncthreads();
        bf16x16 pa = load_frag(&plds[wid][0], 0, 32, 0, lane);
        __syncthreads();

        // O += P @ V ; V^T stored [d, key] so each B column is contiguous
        o0 = wmma_bf16(pa, load_frag(Vb, 0,  SEQ, kb, lane), o0);
        o1 = wmma_bf16(pa, load_frag(Vb, 16, SEQ, kb, lane), o1);
        o2 = wmma_bf16(pa, load_frag(Vb, 32, SEQ, kb, lane), o2);
        o3 = wmma_bf16(pa, load_frag(Vb, 48, SEQ, kb, lane), o3);
    }

    // normalize and store O as [token, h*64+d] bf16 (ready for output GEMM)
    int b = bh >> 4, h = bh & 15;
    #pragma unroll
    for (int r = 0; r < 8; r++) {
        float inv = 1.0f / lrow[r];
        int n = q0 + r + 8 * hi;
        bf16* dst = O + ((size_t)(b * SEQ + n)) * INNER + h * HEAD_DIM;
        dst[col]      = (bf16)(o0[r] * inv);
        dst[col + 16] = (bf16)(o1[r] * inv);
        dst[col + 32] = (bf16)(o2[r] * inv);
        dst[col + 48] = (bf16)(o3[r] * inv);
    }
}

// ---------- kernel 4: output projection  out[m,n] = O[m,:] . w_out[n,:] + b[n] ----------
__global__ __launch_bounds__(256) void out_gemm(const bf16* __restrict__ Ob,
                                                const bf16* __restrict__ wb,
                                                const float* __restrict__ bias,
                                                float* __restrict__ out) {
    int lane = threadIdx.x & 31, wid = threadIdx.x >> 5;
    int wm = wid & 3, wn = wid >> 2;
    int m0 = blockIdx.y * 128 + wm * 32;
    int n0 = blockIdx.x * 128 + wn * 64;

    f32x8 acc[2][4] = {};
    for (int k0 = 0; k0 < INNER; k0 += 32) {
        if (k0 + 32 < INNER) {
            __builtin_prefetch(Ob + (size_t)(m0 + lane) * INNER + k0 + 32, 0, 3);
            __builtin_prefetch(wb + (size_t)(n0 + 2 * lane) * INNER + k0 + 32, 0, 3);
        }
        bf16x16 a0 = load_frag(Ob, m0,      INNER, k0, lane);
        bf16x16 a1 = load_frag(Ob, m0 + 16, INNER, k0, lane);
        bf16x16 b0 = load_frag(wb, n0,      INNER, k0, lane);
        bf16x16 b1 = load_frag(wb, n0 + 16, INNER, k0, lane);
        bf16x16 b2 = load_frag(wb, n0 + 32, INNER, k0, lane);
        bf16x16 b3 = load_frag(wb, n0 + 48, INNER, k0, lane);
        acc[0][0] = wmma_bf16(a0, b0, acc[0][0]);
        acc[0][1] = wmma_bf16(a0, b1, acc[0][1]);
        acc[0][2] = wmma_bf16(a0, b2, acc[0][2]);
        acc[0][3] = wmma_bf16(a0, b3, acc[0][3]);
        acc[1][0] = wmma_bf16(a1, b0, acc[1][0]);
        acc[1][1] = wmma_bf16(a1, b1, acc[1][1]);
        acc[1][2] = wmma_bf16(a1, b2, acc[1][2]);
        acc[1][3] = wmma_bf16(a1, b3, acc[1][3]);
    }
    int hi = lane >> 4, col = lane & 15;
    #pragma unroll
    for (int i = 0; i < 2; i++)
        #pragma unroll
        for (int t = 0; t < 4; t++) {
            int n = n0 + t * 16 + col;
            float bv = bias[n];
            #pragma unroll
            for (int r = 0; r < 8; r++) {
                int m = m0 + i * 16 + r + 8 * hi;
                out[(size_t)m * TOKDIM + n] = acc[i][t][r] + bv;
            }
        }
}

// ---------- workspace layout (bytes) ----------
static const size_t OFF_XB   = 0;                                  // 4096*1024 bf16 = 8 MB
static const size_t OFF_WQKV = OFF_XB   + (size_t)NTOK * TOKDIM * 2;     // 6 MB
static const size_t OFF_WOUT = OFF_WQKV + (size_t)3 * INNER * TOKDIM * 2; // 2 MB
static const size_t OFF_Q    = OFF_WOUT + (size_t)TOKDIM * INNER * 2;     // 8 MB
static const size_t OFF_K    = OFF_Q    + (size_t)NTOK * HEAD_DIM * HEADS * 2;
static const size_t OFF_VT   = OFF_K    + (size_t)NTOK * HEAD_DIM * HEADS * 2;
static const size_t OFF_O    = OFF_VT   + (size_t)NTOK * HEAD_DIM * HEADS * 2;

extern "C" void kernel_launch(void* const* d_in, const int* in_sizes, int n_in,
                              void* d_out, int out_size, void* d_ws, size_t ws_size,
                              hipStream_t stream) {
    const float* x     = (const float*)d_in[0];
    const float* w_qkv = (const float*)d_in[1];
    const float* w_out = (const float*)d_in[2];
    const float* b_out = (const float*)d_in[3];
    float* out = (float*)d_out;

    char* ws = (char*)d_ws;
    bf16* xb    = (bf16*)(ws + OFF_XB);
    bf16* wqkvb = (bf16*)(ws + OFF_WQKV);
    bf16* woutb = (bf16*)(ws + OFF_WOUT);
    bf16* Q     = (bf16*)(ws + OFF_Q);
    bf16* K     = (bf16*)(ws + OFF_K);
    bf16* Vt    = (bf16*)(ws + OFF_VT);
    bf16* O     = (bf16*)(ws + OFF_O);

    // 1) converts
    {
        int n4 = (NTOK * TOKDIM) / 4;
        cvt_f32_bf16<<<(n4 + 255) / 256, 256, 0, stream>>>(x, xb, n4);
        n4 = (3 * INNER * TOKDIM) / 4;
        cvt_f32_bf16<<<(n4 + 255) / 256, 256, 0, stream>>>(w_qkv, wqkvb, n4);
        n4 = (TOKDIM * INNER) / 4;
        cvt_f32_bf16<<<(n4 + 255) / 256, 256, 0, stream>>>(w_out, woutb, n4);
    }
    // 2) QKV projection: [4096 x 1024] * [1024 x 3072]
    qkv_gemm<<<dim3(3 * INNER / 128, NTOK / 128), 256, 0, stream>>>(xb, wqkvb, Q, K, Vt);
    // 3) flash attention: 32 batch-heads x 16 query blocks of 128 rows
    attn_kernel<<<dim3(BATCH * HEADS, SEQ / 128), 256, 0, stream>>>(Q, K, Vt, O);
    // 4) output projection + bias
    out_gemm<<<dim3(TOKDIM / 128, NTOK / 128), 256, 0, stream>>>(O, woutb, b_out, out);
}
